// LigerFusedLinearGRPOLoss_19164144075542
// MI455X (gfx1250) — matrix operand order repaired and checked by reading, not compiled, not run on hardware
//
#include <hip/hip_runtime.h>
#include <hip/hip_bf16.h>
#include <math.h>

// ---------------------------------------------------------------------------
// GRPO fused linear loss for MI455X (gfx1250, wave32, WMMA bf16).
//   logits = X @ W^T + bias   (never materialized)
//   per-token: max log-softmax = max(logits) - logsumexp(logits)
//   seq_logps[b] = sum_s mask * maxlp ;  loss/metrics on B=4 scalars
//
// v6: A-fragment addressing via per-tile 32-bit LDS offsets dereferenced
// directly off the __shared__ array (restores ds_load_b128 instead of the
// flat_load_b128 the generic pointer array produced), with constant bumps so
// all loads keep immediate offsets. 4 M-tiles per workgroup (M=64, 257 KB
// LDS); W pre-converted to bf16 in workspace when it fits, so bf16 W
// (131 MB/model) streams HBM once and lives in the 192 MB L2.
// ---------------------------------------------------------------------------

typedef __attribute__((ext_vector_type(16))) __bf16 v16bf;
typedef __attribute__((ext_vector_type(8)))  __bf16 v8bf;
typedef __attribute__((ext_vector_type(8)))  float  v8f;
typedef __attribute__((ext_vector_type(4)))  float  v4f;

#define B_   4
#define S_   512
#define H_   2048
#define V_   32000
#define TOK  (B_ * S_)          // 2048 tokens
#define MT   4                  // M tiles per workgroup
#define RBG  (TOK / (16 * MT))  // 32 row groups of 64 tokens
#define NSLICES 10
#define SLICE_V (V_ / NSLICES)  // 3200 vocab cols per slice
#define SLICE_TILES (SLICE_V / 16) // 200 tiles of 16 cols
#define WAVES 8
#define TILES_PER_WAVE (SLICE_TILES / WAVES) // 25
#define KSTEPS (H_ / 32)        // 64 WMMA k-steps (bf16 k=32)
#define BETA 0.1f
#define NEG_INF (-3.0e38f)
// Padded LDS layout: row stride 4096+16 B keeps rows on distinct bank groups
// (bank advance = 1028 mod 64 = 4 per row) while keeping addresses linear.
#define ROW_STRIDE 4112
#define TILE_STRIDE (16 * ROW_STRIDE)   // 65792 B per 16 x 2048 bf16 A tile

// ---- W f32 -> bf16 pre-convert (streaming, memory bound) ----
__global__ __launch_bounds__(256)
void cvt_f32_bf16_kernel(const float* __restrict__ src,
                         __bf16* __restrict__ dst, long long n) {
  const long long stride = (long long)gridDim.x * blockDim.x * 8;
  for (long long i = ((long long)blockIdx.x * blockDim.x + threadIdx.x) * 8;
       i < n; i += stride) {
    v4f f0 = *(const v4f*)(src + i);
    v4f f1 = *(const v4f*)(src + i + 4);
    v8bf pk;
    pk[0] = (__bf16)f0[0]; pk[1] = (__bf16)f0[1];
    pk[2] = (__bf16)f0[2]; pk[3] = (__bf16)f0[3];
    pk[4] = (__bf16)f1[0]; pk[5] = (__bf16)f1[1];
    pk[6] = (__bf16)f1[2]; pk[7] = (__bf16)f1[3];
    *(v8bf*)(dst + i) = pk;
  }
}

// Load one lane's 16-element B-fragment slice (16 consecutive K values).
__device__ __forceinline__ v16bf load_bfrag(const float* wp) {
  __builtin_prefetch(wp + 8 * 32, 0, 3);
  v4f b0 = *(const v4f*)(wp + 0);
  v4f b1 = *(const v4f*)(wp + 4);
  v4f b2 = *(const v4f*)(wp + 8);
  v4f b3 = *(const v4f*)(wp + 12);
  v16bf bf;
  bf[0]  = (__bf16)b0[0]; bf[1]  = (__bf16)b0[1];
  bf[2]  = (__bf16)b0[2]; bf[3]  = (__bf16)b0[3];
  bf[4]  = (__bf16)b1[0]; bf[5]  = (__bf16)b1[1];
  bf[6]  = (__bf16)b1[2]; bf[7]  = (__bf16)b1[3];
  bf[8]  = (__bf16)b2[0]; bf[9]  = (__bf16)b2[1];
  bf[10] = (__bf16)b2[2]; bf[11] = (__bf16)b2[3];
  bf[12] = (__bf16)b3[0]; bf[13] = (__bf16)b3[1];
  bf[14] = (__bf16)b3[2]; bf[15] = (__bf16)b3[3];
  return bf;
}
__device__ __forceinline__ v16bf load_bfrag(const __bf16* wp) {
  __builtin_prefetch(wp + 8 * 32, 0, 3);
  v8bf b0 = *(const v8bf*)(wp + 0);
  v8bf b1 = *(const v8bf*)(wp + 8);
  return __builtin_shufflevector(b0, b1, 0, 1, 2, 3, 4, 5, 6, 7,
                                         8, 9, 10, 11, 12, 13, 14, 15);
}

__device__ __forceinline__ v16bf cat8(v8bf a0, v8bf a1) {
  return __builtin_shufflevector(a0, a1, 0, 1, 2, 3, 4, 5, 6, 7,
                                         8, 9, 10, 11, 12, 13, 14, 15);
}

template <typename WT>
__global__ __launch_bounds__(256)
void grpo_logits_kernel(const float* __restrict__ X,     // [TOK, H]
                        const WT* __restrict__ W,        // [V, H]
                        const float* __restrict__ bias,  // [V]
                        float* __restrict__ partMS,      // [NSLICES*TOK*2] (m,s)
                        float* __restrict__ partLS)      // [NSLICES*RBG] logit sums
{
  __shared__ __align__(16) unsigned char lds[MT * TILE_STRIDE]; // 257 KB

  const int rbg   = blockIdx.x;   // row group (fastest -> L2 reuse of W slice)
  const int slice = blockIdx.y;
  const int tid   = threadIdx.x;
  const int lane  = tid & 31;
  const int wave  = tid >> 5;

  // ---- stage MT A tiles (64 tokens x H) as bf16 into LDS ----
  {
    const int row     = tid & 15;
    const int atile   = (tid >> 4) & (MT - 1);
    const int colbase = (tid >> 6) * 512;       // 4 groups x 512 cols
    const float* src =
        X + (size_t)(rbg * 16 * MT + atile * 16 + row) * H_ + colbase;
    const unsigned dbase =
        (unsigned)(atile * TILE_STRIDE + row * ROW_STRIDE + colbase * 2);
#pragma unroll 4
    for (int i = 0; i < 64; ++i) {
      v4f f0 = *(const v4f*)(src + 8 * i);
      v4f f1 = *(const v4f*)(src + 8 * i + 4);
      v8bf pk;
      pk[0] = (__bf16)f0[0]; pk[1] = (__bf16)f0[1];
      pk[2] = (__bf16)f0[2]; pk[3] = (__bf16)f0[3];
      pk[4] = (__bf16)f1[0]; pk[5] = (__bf16)f1[1];
      pk[6] = (__bf16)f1[2]; pk[7] = (__bf16)f1[3];
      *(v8bf*)(lds + dbase + 16 * i) = pk;
    }
  }
  __syncthreads();

  // A-fragment addressing, contiguous k order (same as the B fragment):
  //   lanes 0-15:  M=lane,    K = {0..15}
  //   lanes 16-31: M=lane-16, K = {16..31}
  const int Mrow = lane & 15;
  const int kb   = (lane < 16) ? 0 : 16;
  const unsigned abase = (unsigned)(Mrow * ROW_STRIDE + kb * 2);
  // B-fragment: lane holds one vocab column, 16 consecutive K values.
  const int nofs  = lane & 15;
  const int khalf = (lane < 16) ? 0 : 16;

  float m[MT][8], s[MT][8];
  float lsum = 0.f;
#pragma unroll
  for (int t = 0; t < MT; ++t)
#pragma unroll
    for (int r = 0; r < 8; ++r) { m[t][r] = NEG_INF; s[t][r] = 0.f; }

  for (int tt = 0; tt < TILES_PER_WAVE; ++tt) {
    const int tile = wave + tt * WAVES;
    const int col0 = slice * SLICE_V + tile * 16;
    const WT* wp = W + (size_t)(col0 + nofs) * H_ + khalf;

    v8f acc[MT];
    unsigned aoff[MT];   // 32-bit LDS byte offsets, one induction per tile
#pragma unroll
    for (int t = 0; t < MT; ++t) {
      acc[t]  = (v8f){};
      aoff[t] = abase + (unsigned)(t * TILE_STRIDE);
    }

    // Two k-steps per body: all LDS offsets are constants 0/16/64/80 off the
    // per-tile offset registers, which advance by a constant 128 B.
    for (int k2 = 0; k2 < KSTEPS / 2; ++k2) {
      const v16bf bf0 = load_bfrag(wp);
      const v16bf bf1 = load_bfrag(wp + 32);
      wp += 64;
#pragma unroll
      for (int t = 0; t < MT; ++t) {
        v8bf a0 = *(const v8bf*)(lds + aoff[t] + 0);
        v8bf a1 = *(const v8bf*)(lds + aoff[t] + 16);
        acc[t] = __builtin_amdgcn_wmma_f32_16x16x32_bf16(
            false, cat8(a0, a1), false, bf0, (short)0, acc[t], false, false);
      }
#pragma unroll
      for (int t = 0; t < MT; ++t) {
        v8bf a0 = *(const v8bf*)(lds + aoff[t] + 64);
        v8bf a1 = *(const v8bf*)(lds + aoff[t] + 80);
        acc[t] = __builtin_amdgcn_wmma_f32_16x16x32_bf16(
            false, cat8(a0, a1), false, bf1, (short)0, acc[t], false, false);
        aoff[t] += 128;
      }
    }

    // Online softmax update: one (max, sum-exp) step per finished logit.
    const float bv = bias[col0 + nofs];
#pragma unroll
    for (int t = 0; t < MT; ++t) {
#pragma unroll
      for (int r = 0; r < 8; ++r) {
        const float v = acc[t][r] + bv;
        lsum += v;
        const float mn = fmaxf(m[t][r], v);
        s[t][r] = s[t][r] * __expf(m[t][r] - mn) + __expf(v - mn);
        m[t][r] = mn;
      }
    }
  }
  __syncthreads();  // A tiles no longer needed; LDS reused for merge buffers

  // Merge (m,s) across the 16 N-lanes of each half-wave (xor butterfly).
#pragma unroll
  for (int msk = 1; msk <= 8; msk <<= 1) {
#pragma unroll
    for (int t = 0; t < MT; ++t) {
#pragma unroll
      for (int r = 0; r < 8; ++r) {
        const float mo = __shfl_xor(m[t][r], msk, 32);
        const float so = __shfl_xor(s[t][r], msk, 32);
        const float mn = fmaxf(m[t][r], mo);
        s[t][r] = s[t][r] * __expf(m[t][r] - mn) + so * __expf(mo - mn);
        m[t][r] = mn;
      }
    }
  }
  // Full-wave reduce of the raw logit sum.
#pragma unroll
  for (int msk = 1; msk <= 16; msk <<= 1) lsum += __shfl_xor(lsum, msk, 32);

  float2* wgMS = (float2*)lds;             // [8 waves][64 rows]
  float*  wgLS = (float*)(lds + 16384);    // [8 waves]
  if (lane == 0 || lane == 16) {
    const int rbase = (lane == 0) ? 0 : 8; // VGPR r <-> M=r (lanes<16) / M=r+8
#pragma unroll
    for (int t = 0; t < MT; ++t)
#pragma unroll
      for (int r = 0; r < 8; ++r)
        wgMS[wave * (16 * MT) + t * 16 + rbase + r] =
            make_float2(m[t][r], s[t][r]);
  }
  if (lane == 0) wgLS[wave] = lsum;
  __syncthreads();

  if (tid < 16 * MT) {  // merge the 8 waves' disjoint N ranges per token row
    float mm = wgMS[tid].x, ss = wgMS[tid].y;
    for (int w = 1; w < 8; ++w) {
      const float2 o = wgMS[w * (16 * MT) + tid];
      const float mn = fmaxf(mm, o.x);
      ss = ss * __expf(mm - mn) + o.y * __expf(o.x - mn);
      mm = mn;
    }
    const size_t idx = (size_t)slice * TOK + rbg * (16 * MT) + tid;
    partMS[idx * 2 + 0] = mm;
    partMS[idx * 2 + 1] = ss;
  }
  if (tid == 0) {
    float t = 0.f;
    for (int w = 0; w < 8; ++w) t += wgLS[w];
    partLS[slice * RBG + rbg] = t;
  }
}

__global__ __launch_bounds__(256)
void grpo_finalize_kernel(const float* __restrict__ pMS,  // policy (m,s)
                          const float* __restrict__ pLS,  // policy logit sums
                          const float* __restrict__ rMS,  // ref (m,s)
                          const float* __restrict__ mask, // [B,S]
                          const float* __restrict__ rewards, // [B]
                          float* __restrict__ out)        // [5]
{
  __shared__ float seqP[B_], seqR[B_], lsumSh;
  if (threadIdx.x < B_) { seqP[threadIdx.x] = 0.f; seqR[threadIdx.x] = 0.f; }
  if (threadIdx.x == 0) lsumSh = 0.f;
  __syncthreads();

  for (int t = threadIdx.x; t < TOK; t += 256) {
    float mmP = NEG_INF, ssP = 0.f, mmR = NEG_INF, ssR = 0.f;
    for (int sl = 0; sl < NSLICES; ++sl) {
      const size_t idx = (size_t)sl * TOK + t;
      {
        const float m2 = pMS[2 * idx], s2 = pMS[2 * idx + 1];
        const float mn = fmaxf(mmP, m2);
        ssP = ssP * __expf(mmP - mn) + s2 * __expf(m2 - mn);
        mmP = mn;
      }
      {
        const float m2 = rMS[2 * idx], s2 = rMS[2 * idx + 1];
        const float mn = fmaxf(mmR, m2);
        ssR = ssR * __expf(mmR - mn) + s2 * __expf(m2 - mn);
        mmR = mn;
      }
    }
    const float lpP = -logf(ssP);  // max - (max + log(sum)) = -log(sum)
    const float lpR = -logf(ssR);
    const int b = t / S_;
    const float mk = mask[t];
    atomicAdd(&seqP[b], lpP * mk);
    atomicAdd(&seqR[b], lpR * mk);
  }

  float ls = 0.f;
  for (int i = threadIdx.x; i < NSLICES * RBG; i += 256) ls += pLS[i];
#pragma unroll
  for (int msk = 1; msk <= 16; msk <<= 1) ls += __shfl_xor(ls, msk, 32);
  if ((threadIdx.x & 31) == 0) atomicAdd(&lsumSh, ls);
  __syncthreads();

  if (threadIdx.x == 0) {
    float rm = 0.f;
    for (int b = 0; b < B_; ++b) rm += rewards[b];
    rm /= (float)B_;
    float var = 0.f;
    for (int b = 0; b < B_; ++b) {
      const float d = rewards[b] - rm;
      var += d * d;
    }
    const float rstd = sqrtf(var / (float)(B_ - 1));

    float loss = 0.f, seqMean = 0.f, klMean = 0.f;
    for (int b = 0; b < B_; ++b) {
      float adv = rewards[b] - rm;
      if (rstd > 0.f) adv /= rstd;
      const float kl = seqP[b] - seqR[b];
      loss += -(adv * seqP[b]) + BETA * kl;
      seqMean += seqP[b];
      klMean += kl;
    }
    loss /= (float)B_; seqMean /= (float)B_; klMean /= (float)B_;
    float sv = 0.f;
    for (int b = 0; b < B_; ++b) {
      const float d = seqP[b] - seqMean;
      sv += d * d;
    }
    const float seqStd = sqrtf(sv / (float)(B_ - 1));
    out[0] = loss;
    out[1] = seqMean;
    out[2] = seqStd;
    out[3] = lsumSh / ((float)TOK * (float)V_);
    out[4] = klMean;
  }
}

extern "C" void kernel_launch(void* const* d_in, const int* in_sizes, int n_in,
                              void* d_out, int out_size, void* d_ws, size_t ws_size,
                              hipStream_t stream) {
  const float* W     = (const float*)d_in[0]; // lin_weight [V,H]
  const float* X     = (const float*)d_in[1]; // _input [B,S,H]
  const float* mask  = (const float*)d_in[2]; // [B,S]
  const float* rew   = (const float*)d_in[3]; // [B]
  const float* bias  = (const float*)d_in[4]; // [V]
  const float* Xr    = (const float*)d_in[5]; // ref_input
  const float* Wr    = (const float*)d_in[6]; // ref_weight
  const float* biasr = (const float*)d_in[7]; // ref_bias
  (void)in_sizes; (void)n_in; (void)out_size;

  // Workspace layout (floats): partials first, then optional bf16 weights.
  const size_t nMS = (size_t)NSLICES * TOK * 2;  // 40960 per model
  const size_t nLS = (size_t)NSLICES * RBG;      // 320 per model
  float* pMS = (float*)d_ws;
  float* pLS = pMS + nMS;
  float* rMS = pLS + nLS;
  float* rLS = rMS + nMS;
  const size_t partFloats = 2 * (nMS + nLS);
  const size_t partBytes  = ((partFloats * 4 + 255) / 256) * 256; // align 256B
  const size_t wElems     = (size_t)V_ * H_;
  const size_t needPrecvt = partBytes + 2 * wElems * sizeof(__bf16);

  dim3 grid(RBG, NSLICES);
  if (ws_size >= needPrecvt) {
    __bf16* wbf  = (__bf16*)((char*)d_ws + partBytes);
    __bf16* wbfR = wbf + wElems;
    cvt_f32_bf16_kernel<<<2048, 256, 0, stream>>>(W,  wbf,  (long long)wElems);
    cvt_f32_bf16_kernel<<<2048, 256, 0, stream>>>(Wr, wbfR, (long long)wElems);
    grpo_logits_kernel<__bf16><<<grid, 256, 0, stream>>>(X,  wbf,  bias,  pMS, pLS);
    grpo_logits_kernel<__bf16><<<grid, 256, 0, stream>>>(Xr, wbfR, biasr, rMS, rLS);
  } else {
    grpo_logits_kernel<float><<<grid, 256, 0, stream>>>(X,  W,  bias,  pMS, pLS);
    grpo_logits_kernel<float><<<grid, 256, 0, stream>>>(Xr, Wr, biasr, rMS, rLS);
  }
  (void)rLS;
  grpo_finalize_kernel<<<1, 256, 0, stream>>>(pMS, pLS, rMS, mask, rew,
                                              (float*)d_out);
}
